// HandshakingKernel_31026843746815
// MI455X (gfx1250) — matrix operand order repaired and checked by reading, not compiled, not run on hardware
//
#include <hip/hip_runtime.h>
#include <hip/hip_bf16.h>
#include <math.h>

// Problem constants (B, L, H) = (2, 384, 768)
#define LSEQ   384
#define BB     2
#define HDIM   768
#define H2     1536
#define ROWS   (BB * LSEQ)                 // 768 rows of X (B*L, H)
#define NPAIRS ((LSEQ * (LSEQ + 1)) / 2)   // 73920 triu pairs

typedef __attribute__((ext_vector_type(16))) __bf16 v16bf;
typedef __attribute__((ext_vector_type(8)))  float  v8f;
typedef __attribute__((ext_vector_type(4)))  float  v4f;

// Native CDNA5 V_TANH_F32 if the toolchain exposes it; else libm.
#if defined(__has_builtin)
#if __has_builtin(__builtin_amdgcn_tanhf)
#define FAST_TANHF(x) __builtin_amdgcn_tanhf(x)
#endif
#endif
#ifndef FAST_TANHF
#define FAST_TANHF(x) tanhf(x)
#endif

// ---------------------------------------------------------------------------
// Kernel 1: P[row,h] = sum_k X[row,k] * W[h,k]        (left half of W)
//           Q[row,h] = sum_k X[row,k] * W[h,H+k]      (right half of W)
// One wave computes a 16x64 output tile (A-frag reused across 4 WMMA n-tiles).
// f32 inputs are split into bf16 hi+lo; 3 WMMAs/tile/step recover ~f32 accuracy.
// ---------------------------------------------------------------------------
__global__ __launch_bounds__(256) void proj_wmma_kernel(
    const float* __restrict__ X, const float* __restrict__ W,
    float* __restrict__ P, float* __restrict__ Q) {
  const int lane = threadIdx.x & 31;
  const int half = lane >> 4;      // which 16-lane group
  const int r    = lane & 15;

  // 48 M-tiles x 24 N64-groups = 1152 waves; 8 waves/block -> 144 blocks
  const int wid = blockIdx.x * (blockDim.x >> 5) + (threadIdx.x >> 5);
  const int tm  = wid % 48;        // 16-row tile of X
  const int tn  = wid / 48;        // 64-col group over the 1536 output cols
  const int colBase = tn * 64;
  const int koff  = (colBase >= HDIM) ? HDIM : 0;   // which half of W's columns
  float* outMat   = (colBase >= HDIM) ? Q : P;
  const int cbase = colBase - koff;                 // col within P or Q

  const float* arow = X + (size_t)(tm * 16 + r) * HDIM;

  const v8f vzero = {0.f, 0.f, 0.f, 0.f, 0.f, 0.f, 0.f, 0.f};
  v8f acc[4];
#pragma unroll
  for (int s = 0; s < 4; ++s) acc[s] = vzero;

  for (int kb = 0; kb < HDIM; kb += 32) {
    // ---- A fragment (16x32 bf16 layout, ISA 7.12.2):
    // lane-half 0 holds K in {kb+0..7, kb+16..23}; half 1 holds {kb+8..15, kb+24..31}
    const int run1 = kb + 8 * half;
    const int run2 = kb + 16 + 8 * half;
    float af[16];
    *(v4f*)(af + 0)  = *(const v4f*)(arow + run1);
    *(v4f*)(af + 4)  = *(const v4f*)(arow + run1 + 4);
    *(v4f*)(af + 8)  = *(const v4f*)(arow + run2);
    *(v4f*)(af + 12) = *(const v4f*)(arow + run2 + 4);
    v16bf a_hi, a_lo;
#pragma unroll
    for (int e = 0; e < 16; ++e) {
      float x  = af[e];
      __bf16 h = (__bf16)x;
      a_hi[e]  = h;
      a_lo[e]  = (__bf16)(x - (float)h);
    }

#pragma unroll
    for (int s = 0; s < 4; ++s) {
      // ---- B fragment (32x16 bf16): N = r, K = {8*half+0..7, 16+8*half+0..7}
      const float* brow =
          W + (size_t)(cbase + s * 16 + r) * H2 + koff + kb + 8 * half;
      float bf[16];
      *(v4f*)(bf + 0)  = *(const v4f*)(brow + 0);
      *(v4f*)(bf + 4)  = *(const v4f*)(brow + 4);
      *(v4f*)(bf + 8)  = *(const v4f*)(brow + 16);
      *(v4f*)(bf + 12) = *(const v4f*)(brow + 20);
      v16bf b_hi, b_lo;
#pragma unroll
      for (int e = 0; e < 16; ++e) {
        float x  = bf[e];
        __bf16 h = (__bf16)x;
        b_hi[e]  = h;
        b_lo[e]  = (__bf16)(x - (float)h);
      }
      // D = A*B + C  (hi*hi + hi*lo + lo*hi ~ full f32 product)
      acc[s] = __builtin_amdgcn_wmma_f32_16x16x32_bf16(
          false, a_hi, false, b_hi, (short)0, acc[s], false, false);
      acc[s] = __builtin_amdgcn_wmma_f32_16x16x32_bf16(
          false, a_hi, false, b_lo, (short)0, acc[s], false, false);
      acc[s] = __builtin_amdgcn_wmma_f32_16x16x32_bf16(
          false, a_lo, false, b_hi, (short)0, acc[s], false, false);
    }
  }

  // ---- C/D store: VGPR v holds row M = v + 8*half, col N = r
#pragma unroll
  for (int s = 0; s < 4; ++s) {
    const int col = cbase + s * 16 + r;
#pragma unroll
    for (int v = 0; v < 8; ++v) {
      const int row = tm * 16 + v + 8 * half;
      outMat[(size_t)row * HDIM + col] = acc[s][v];
    }
  }
}

// ---------------------------------------------------------------------------
// Kernel 2: out[b,m,h] = tanh(P[b,i(m),h] + Q[b,j(m),h] + bias[h])
// One block per (b,m) row; 192 threads x float4 = 768 floats.
// P/Q/bias are L2-hot (4.7 MB); output stores are non-temporal (454 MB stream).
// ---------------------------------------------------------------------------
__global__ __launch_bounds__(192) void pair_tanh_kernel(
    const float* __restrict__ P, const float* __restrict__ Q,
    const float* __restrict__ bias, float* __restrict__ out) {
  const int bm = blockIdx.x;               // 0 .. B*NPAIRS-1
  const int b  = bm / NPAIRS;
  const int m  = bm - b * NPAIRS;

  // Invert row-major triu index: rowStart(i) = i*L - i*(i-1)/2
  const float tf = 2.0f * LSEQ + 1.0f;
  int i = (int)((tf - sqrtf(tf * tf - 8.0f * (float)m)) * 0.5f);
  if (i < 0) i = 0;
  if (i > LSEQ - 1) i = LSEQ - 1;
  while (((i + 1) * LSEQ - ((i + 1) * i) / 2) <= m) ++i;
  while ((i * LSEQ - (i * (i - 1)) / 2) > m) --i;
  const int j = m - (i * LSEQ - (i * (i - 1)) / 2) + i;

  const v4f* prow = (const v4f*)(P + (size_t)(b * LSEQ + i) * HDIM);
  const v4f* qrow = (const v4f*)(Q + (size_t)(b * LSEQ + j) * HDIM);
  const v4f* brow = (const v4f*)bias;
  v4f* orow       = (v4f*)(out + (size_t)bm * HDIM);

  const int t = threadIdx.x;               // 0..191
  const v4f p  = prow[t];
  const v4f q  = qrow[t];
  const v4f bb = brow[t];
  v4f o;
  o.x = FAST_TANHF(p.x + q.x + bb.x);
  o.y = FAST_TANHF(p.y + q.y + bb.y);
  o.z = FAST_TANHF(p.z + q.z + bb.z);
  o.w = FAST_TANHF(p.w + q.w + bb.w);
  __builtin_nontemporal_store(o, orow + t);
}

extern "C" void kernel_launch(void* const* d_in, const int* in_sizes, int n_in,
                              void* d_out, int out_size, void* d_ws,
                              size_t ws_size, hipStream_t stream) {
  (void)in_sizes; (void)n_in; (void)out_size; (void)ws_size;
  const float* X    = (const float*)d_in[0];  // (B, L, H) f32
  const float* W    = (const float*)d_in[1];  // (H, 2H) f32
  const float* bias = (const float*)d_in[2];  // (H,) f32
  float* out = (float*)d_out;                 // (B, M, H) f32

  float* P = (float*)d_ws;                    // (B*L, H)
  float* Q = P + (size_t)ROWS * HDIM;         // (B*L, H)

  // 1152 waves / 8 waves-per-block = 144 blocks
  proj_wmma_kernel<<<144, 256, 0, stream>>>(X, W, P, Q);
  // One block per output pair-row
  pair_tanh_kernel<<<BB * NPAIRS, 192, 0, stream>>>(P, Q, bias, out);
}
